// LinearAttention_86431921865156
// MI455X (gfx1250) — matrix operand order repaired and checked by reading, not compiled, not run on hardware
//
#include <hip/hip_runtime.h>
#include <hip/hip_bf16.h>

// MI455X (gfx1250) implementation: bf16 WMMA pipeline + TDM async V staging.
// Shapes fixed by the reference: B=2, N=2048, D=1024, H=16, HD=64.

#define BQ     2
#define NSEQ   2048
#define DMODEL 1024
#define NH     16
#define HD     64

typedef __bf16 bf16;
typedef __attribute__((ext_vector_type(16))) __bf16 v16bf;
typedef __attribute__((ext_vector_type(8)))  __bf16 v8bf;
typedef __attribute__((ext_vector_type(8)))  float  v8f;
typedef __attribute__((ext_vector_type(4)))  unsigned int v4u;
typedef __attribute__((ext_vector_type(8)))  int v8i;
typedef __attribute__((ext_vector_type(4)))  int v4i;

// D = A(16x32) * B(32x16) + C, bf16 inputs, f32 accumulate.
__device__ __forceinline__ v8f wmma_bf16(v16bf a, v16bf b, v8f c) {
  return __builtin_amdgcn_wmma_f32_16x16x32_bf16(
      /*neg_a=*/false, a, /*neg_b=*/false, b,
      /*c_mod=*/(short)0, c, /*reuse_a=*/false, /*reuse_b=*/false);
}

// ---- Fragment loaders (per CDNA5 ISA 7.12.2, wave32) ----
// 16-bit A 16x32 (MxK): lane L<16 holds row M=L, K = 0..7 then 16..23;
// lane L+16 holds row M=L, K = 8..15 then 24..31.
__device__ __forceinline__ v16bf load_a_g(const bf16* p0, int lda) {
  int lane = threadIdx.x & 31;
  int m = lane & 15, hf = lane >> 4;
  const bf16* p = p0 + (size_t)m * lda + hf * 8;
  union { v16bf v; v8bf h[2]; } u;
  u.h[0] = *(const v8bf*)(p);
  u.h[1] = *(const v8bf*)(p + 16);
  return u.v;
}

// 16-bit B 32x16 (KxN) of W^T where W is [N,K] row-major:
// lane L<16 holds column N=L with K = 0..7 then 16..23; lane L+16 holds K = 8..15 then 24..31.
__device__ __forceinline__ v16bf load_b_g(const bf16* p0, int ldw) {
  int lane = threadIdx.x & 31;
  int n = lane & 15, hf = lane >> 4;
  const bf16* p = p0 + (size_t)n * ldw + hf * 8;
  union { v16bf v; v8bf h[2]; } u;
  u.h[0] = *(const v8bf*)(p);
  u.h[1] = *(const v8bf*)(p + 16);
  return u.v;
}

// ---- Tensor Data Mover: 2D bf16 tile (rows x cols, row stride ld) -> LDS ----
// D# layout per CDNA5 ISA (group0 + group1, 2D).
// This toolchain exposes the 6-arg builtin:
//   (uint32x4 g0, int32x8 g1, int32x4 g2, int32x4 g3, int32x8 pad, i32 cpol)
__device__ __forceinline__ void tdm_load_2d(unsigned int lds_off, const bf16* gptr,
                                            int rows, int cols, int ld) {
  unsigned long long ga = (unsigned long long)(uintptr_t)gptr;
  v4u g0;
  g0[0] = 1u;                                   // count=1, user descriptor
  g0[1] = lds_off;                              // lds_addr (bytes)
  g0[2] = (unsigned int)ga;                     // global_addr[31:0]
  g0[3] = (unsigned int)((ga >> 32) & 0x01FFFFFFu) | (2u << 30);  // addr hi | type=2
  unsigned int dim0 = (unsigned int)cols;       // tensor sized exactly = tile
  unsigned int dim1 = (unsigned int)rows;
  unsigned long long st0 = (unsigned long long)ld;  // dim0 stride (elements)
  v8i g1;
  g1[0] = (int)(1u << 16);                      // data_size=1 (2 bytes/elem)
  g1[1] = (int)((dim0 & 0xFFFFu) << 16);        // tensor_dim0[15:0]
  g1[2] = (int)((dim0 >> 16) | ((dim1 & 0xFFFFu) << 16));   // dim0 hi | dim1 lo
  g1[3] = (int)((dim1 >> 16) | ((unsigned int)cols << 16)); // dim1 hi | tile_dim0
  g1[4] = (int)((unsigned int)rows & 0xFFFFu);  // tile_dim1 (tile_dim2=0)
  g1[5] = (int)(unsigned int)(st0 & 0xFFFFFFFFu);
  g1[6] = (int)(unsigned int)(st0 >> 32);       // stride hi16; dim1_stride lo16=0
  g1[7] = 0;
  v4i z4 = {0, 0, 0, 0};                        // groups 2/3 unused (2D tensor)
  v8i z8 = {0, 0, 0, 0, 0, 0, 0, 0};
  __builtin_amdgcn_tensor_load_to_lds(g0, g1, z4, z4, z8, 0);
}

// ---- fp32 -> bf16 cast kernel ----
__global__ __launch_bounds__(256) void cast_kernel(const float* __restrict__ in,
                                                   bf16* __restrict__ out, int n) {
  int i = (blockIdx.x * 256 + threadIdx.x) * 4;
  if (i < n) {
    float4 v = *(const float4*)(in + i);
    out[i + 0] = (bf16)v.x;
    out[i + 1] = (bf16)v.y;
    out[i + 2] = (bf16)v.z;
    out[i + 3] = (bf16)v.w;
  }
}

// ---- Shared GEMM core: 32(M) x 64(N) per wave, double-buffered K pipeline ----
__device__ __forceinline__ void gemm_load(const bf16* A, const bf16* W, int m0,
                                          int n0, int k0, int K, v16bf a[2],
                                          v16bf b[4]) {
  a[0] = load_a_g(A + (size_t)m0 * K + k0, K);
  a[1] = load_a_g(A + (size_t)(m0 + 16) * K + k0, K);
#pragma unroll
  for (int t = 0; t < 4; ++t)
    b[t] = load_b_g(W + (size_t)(n0 + t * 16) * K + k0, K);
}

__device__ __forceinline__ void gemm_core(const bf16* A, const bf16* W, int m0,
                                          int n0, int K, v8f acc[2][4]) {
  v16bf a0[2], b0[4], a1[2], b1[4];
  gemm_load(A, W, m0, n0, 0, K, a0, b0);
  for (int k0 = 0; k0 < K; k0 += 64) {
    gemm_load(A, W, m0, n0, k0 + 32, K, a1, b1);  // prefetch next chunk
    __builtin_prefetch(A + (size_t)m0 * K + k0 + 256, 0, 1);
#pragma unroll
    for (int mi = 0; mi < 2; ++mi)
#pragma unroll
      for (int t = 0; t < 4; ++t) acc[mi][t] = wmma_bf16(a0[mi], b0[t], acc[mi][t]);
    if (k0 + 64 < K) gemm_load(A, W, m0, n0, k0 + 64, K, a0, b0);
#pragma unroll
    for (int mi = 0; mi < 2; ++mi)
#pragma unroll
      for (int t = 0; t < 4; ++t) acc[mi][t] = wmma_bf16(a1[mi], b1[t], acc[mi][t]);
  }
}

// ---- QKV projection: [M,K] x [3D,K]^T, silu on q,k in the epilogue ----
// Block = 8 waves (4M x 2N), block tile 128 x 128.
__global__ __launch_bounds__(256) void gemm_qkv_kernel(
    const bf16* __restrict__ A, const bf16* __restrict__ W,
    bf16* __restrict__ qo, bf16* __restrict__ ko, bf16* __restrict__ vo, int K) {
  int wave = threadIdx.x >> 5, lane = threadIdx.x & 31;
  int m0 = blockIdx.x * 128 + (wave & 3) * 32;
  int n0 = blockIdx.y * 128 + (wave >> 2) * 64;
  v8f acc[2][4] = {};
  gemm_core(A, W, m0, n0, K, acc);
  int hf = lane >> 4, cc = lane & 15;
#pragma unroll
  for (int mi = 0; mi < 2; ++mi)
#pragma unroll
    for (int t = 0; t < 4; ++t) {
      int n = n0 + t * 16 + cc;
      int seg = n >> 10;  // 0:q 1:k 2:v (tiles never straddle a segment)
      int d = n & (DMODEL - 1);
      bf16* dst = (seg == 0) ? qo : (seg == 1) ? ko : vo;
#pragma unroll
      for (int r = 0; r < 8; ++r) {
        int m = m0 + mi * 16 + r + 8 * hf;
        float v = acc[mi][t][r];
        if (seg < 2) v = v / (1.f + __expf(-v));  // silu
        dst[(size_t)m * DMODEL + d] = (bf16)v;
      }
    }
}

// ---- Output projection: f32 epilogue ----
__global__ __launch_bounds__(256) void gemm_out_kernel(
    const bf16* __restrict__ A, const bf16* __restrict__ W,
    float* __restrict__ out, int N, int K) {
  int wave = threadIdx.x >> 5, lane = threadIdx.x & 31;
  int m0 = blockIdx.x * 128 + (wave & 3) * 32;
  int n0 = blockIdx.y * 128 + (wave >> 2) * 64;
  v8f acc[2][4] = {};
  gemm_core(A, W, m0, n0, K, acc);
  int hf = lane >> 4, cc = lane & 15;
#pragma unroll
  for (int mi = 0; mi < 2; ++mi)
#pragma unroll
    for (int t = 0; t < 4; ++t)
#pragma unroll
      for (int r = 0; r < 8; ++r)
        out[(size_t)(m0 + mi * 16 + r + 8 * hf) * N + n0 + t * 16 + cc] =
            acc[mi][t][r];
}

// ---- Fused gate MLP: g = sigmoid((x Wg1^T + b1) Wg2^T + b2) ----
__global__ __launch_bounds__(256) void gate_kernel(
    const bf16* __restrict__ xb, const bf16* __restrict__ wg1,
    const float* __restrict__ bg1, const bf16* __restrict__ wg2,
    const float* __restrict__ bg2, float* __restrict__ gate) {
  __shared__ alignas(16) bf16 hl[8][16][HD];
  int wave = threadIdx.x >> 5, lane = threadIdx.x & 31;
  int row0 = (blockIdx.x * 8 + wave) * 16;
  int hf = lane >> 4, cc = lane & 15, m = lane & 15;

  // Stage 1: h[16,64] = x Wg1^T
  v8f hacc[4] = {};
  for (int k0 = 0; k0 < DMODEL; k0 += 32) {
    v16bf a = load_a_g(xb + (size_t)row0 * DMODEL + k0, DMODEL);
#pragma unroll
    for (int t = 0; t < 4; ++t) {
      v16bf b = load_b_g(wg1 + (size_t)(t * 16) * DMODEL + k0, DMODEL);
      hacc[t] = wmma_bf16(a, b, hacc[t]);
    }
  }
#pragma unroll
  for (int t = 0; t < 4; ++t)
#pragma unroll
    for (int r = 0; r < 8; ++r)
      hl[wave][r + 8 * hf][t * 16 + cc] = (bf16)(hacc[t][r] + bg1[t * 16 + cc]);

  // Stage 2: g[16,1024] = sigmoid(h Wg2^T + b2); A fragments from LDS.
  v16bf a0, a1;
#pragma unroll
  for (int e = 0; e < 8; ++e) {
    a0[e]     = hl[wave][m][hf * 8 + e];
    a0[e + 8] = hl[wave][m][16 + hf * 8 + e];
    a1[e]     = hl[wave][m][32 + hf * 8 + e];
    a1[e + 8] = hl[wave][m][48 + hf * 8 + e];
  }
  for (int nt = 0; nt < DMODEL / 16; ++nt) {
    v8f acc = {};
    v16bf b0 = load_b_g(wg2 + (size_t)(nt * 16) * HD + 0, HD);
    v16bf b1 = load_b_g(wg2 + (size_t)(nt * 16) * HD + 32, HD);
    acc = wmma_bf16(a0, b0, acc);
    acc = wmma_bf16(a1, b1, acc);
    float bb = bg2[nt * 16 + cc];
#pragma unroll
    for (int r = 0; r < 8; ++r) {
      float v = acc[r] + bb;
      gate[(size_t)(row0 + r + 8 * hf) * DMODEL + nt * 16 + cc] =
          1.f / (1.f + __expf(-v));
    }
  }
}

// ---- Causal linear attention: O = (QK^T ∘ tril) V per (b,h) ----
// grid = (B*H, 16); wave w handles query tile qt = blockIdx.y + 16*w.
// V tiles are streamed global->LDS by the Tensor Data Mover, double-buffered.
__global__ __launch_bounds__(256) void attn_kernel(
    const bf16* __restrict__ qg, const bf16* __restrict__ kg,
    const bf16* __restrict__ vg, float* __restrict__ outf) {
  __shared__ alignas(16) bf16 sS[8][16][32];
  __shared__ alignas(16) bf16 sV[8][2][32][HD];
  int wave = threadIdx.x >> 5, lane = threadIdx.x & 31;
  int bh = blockIdx.x;
  int b = bh >> 4, h = bh & 15;
  int qt = blockIdx.y + 16 * wave;  // 0..127
  size_t base = (size_t)b * NSEQ * DMODEL + (size_t)h * HD;
  const bf16* Q = qg + base;
  const bf16* Kp = kg + base;
  const bf16* Vp = vg + base;
  int hf = lane >> 4, cc = lane & 15, m = lane & 15;
  int qr0 = qt * 16;

  unsigned int vbase0 = (unsigned int)(uintptr_t)&sV[wave][0][0][0];
  unsigned int vbase1 = (unsigned int)(uintptr_t)&sV[wave][1][0][0];

  v16bf aq0 = load_a_g(Q + (size_t)qr0 * DMODEL + 0, DMODEL);
  v16bf aq1 = load_a_g(Q + (size_t)qr0 * DMODEL + 32, DMODEL);
  v8f oacc[4] = {};

  int npair = (qt >> 1) + 1;  // key-tile pairs (32 keys each)
  // Prologue: TDM-load V pair 0 into buffer 0.
  tdm_load_2d(vbase0, Vp, 32, HD, DMODEL);

  for (int j2 = 0; j2 < npair; ++j2) {
    int cur = j2 & 1;
    // Stream the next V tile while computing this pair.
    if (j2 + 1 < npair)
      tdm_load_2d(cur ? vbase0 : vbase1, Vp + (size_t)(j2 + 1) * 32 * DMODEL, 32,
                  HD, DMODEL);

    // S = Q K^T for two 16-wide key tiles -> masked bf16 into sS[16][32].
#pragma unroll
    for (int s = 0; s < 2; ++s) {
      int jt = 2 * j2 + s;
      if (jt <= qt) {
        v8f sacc = {};
        v16bf bk0 = load_b_g(Kp + (size_t)(jt * 16) * DMODEL + 0, DMODEL);
        v16bf bk1 = load_b_g(Kp + (size_t)(jt * 16) * DMODEL + 32, DMODEL);
        sacc = wmma_bf16(aq0, bk0, sacc);
        sacc = wmma_bf16(aq1, bk1, sacc);
        bool diag = (jt == qt);
#pragma unroll
        for (int r = 0; r < 8; ++r) {
          int rr = r + 8 * hf;
          float v = (!diag || (cc <= rr)) ? sacc[r] : 0.f;  // tril mask
          sS[wave][rr][s * 16 + cc] = (bf16)v;
        }
      } else {
#pragma unroll
        for (int r = 0; r < 8; ++r) sS[wave][r + 8 * hf][s * 16 + cc] = (bf16)0.f;
      }
    }

    // Wait for this pair's V tile (keep the next one in flight).
    if (j2 + 1 < npair)
      __builtin_amdgcn_s_wait_tensorcnt(1);
    else
      __builtin_amdgcn_s_wait_tensorcnt(0);

    // O += S(16x32) * V(32x64): A from sS, B columns gathered from sV.
    v16bf as;
#pragma unroll
    for (int e = 0; e < 8; ++e) {
      as[e]     = sS[wave][m][hf * 8 + e];
      as[e + 8] = sS[wave][m][16 + hf * 8 + e];
    }
#pragma unroll
    for (int t = 0; t < 4; ++t) {
      v16bf bv;
#pragma unroll
      for (int e = 0; e < 16; ++e) bv[e] = sV[wave][cur][(e < 8 ? hf * 8 + e : 8 + hf * 8 + e)][t * 16 + cc];
      oacc[t] = wmma_bf16(as, bv, oacc[t]);
    }
  }
#pragma unroll
  for (int t = 0; t < 4; ++t)
#pragma unroll
    for (int r = 0; r < 8; ++r)
      outf[((size_t)b * NSEQ + qr0 + r + 8 * hf) * DMODEL + h * HD + t * 16 + cc] =
          oacc[t][r];
}

// ---- LayerNorm(out * gate) -> bf16 ----
__global__ __launch_bounds__(256) void ln_kernel(
    const float* __restrict__ attnf, const float* __restrict__ gate,
    const float* __restrict__ gamma, const float* __restrict__ beta,
    bf16* __restrict__ y) {
  __shared__ float red[16];
  int row = blockIdx.x;
  const float* o = attnf + (size_t)row * DMODEL;
  const float* g = gate + (size_t)row * DMODEL;
  float vals[4], s = 0.f, s2 = 0.f;
#pragma unroll
  for (int i = 0; i < 4; ++i) {
    int c = threadIdx.x + i * 256;
    float v = o[c] * g[c];
    vals[i] = v; s += v; s2 += v * v;
  }
#pragma unroll
  for (int off = 16; off; off >>= 1) {
    s  += __shfl_down(s, off, 32);
    s2 += __shfl_down(s2, off, 32);
  }
  int wave = threadIdx.x >> 5, lane = threadIdx.x & 31;
  if (lane == 0) { red[wave] = s; red[8 + wave] = s2; }
  __syncthreads();
  if (threadIdx.x == 0) {
    float ts = 0.f, ts2 = 0.f;
    for (int w = 0; w < 8; ++w) { ts += red[w]; ts2 += red[8 + w]; }
    red[0] = ts; red[8] = ts2;
  }
  __syncthreads();
  float mu = red[0] / DMODEL;
  float var = red[8] / DMODEL - mu * mu;
  float rs = rsqrtf(var + 1e-5f);
#pragma unroll
  for (int i = 0; i < 4; ++i) {
    int c = threadIdx.x + i * 256;
    y[(size_t)row * DMODEL + c] = (bf16)((vals[i] - mu) * rs * gamma[c] + beta[c]);
  }
}

extern "C" void kernel_launch(void* const* d_in, const int* in_sizes, int n_in,
                              void* d_out, int out_size, void* d_ws, size_t ws_size,
                              hipStream_t stream) {
  const float* x     = (const float*)d_in[0];
  const float* wqkv  = (const float*)d_in[1];
  const float* wg1   = (const float*)d_in[2];
  const float* bg1   = (const float*)d_in[3];
  const float* wg2   = (const float*)d_in[4];
  const float* bg2   = (const float*)d_in[5];
  const float* wout  = (const float*)d_in[6];
  const float* gamma = (const float*)d_in[7];
  const float* beta  = (const float*)d_in[8];
  float* out = (float*)d_out;

  const int M = BQ * NSEQ;  // 4096 rows
  char* ws = (char*)d_ws;
  size_t off = 0;
  auto alloc = [&](size_t bytes) -> void* {
    void* p = ws + off;
    off += (bytes + 255) & ~(size_t)255;
    return p;
  };
  bf16* xb    = (bf16*)alloc((size_t)M * DMODEL * 2);
  bf16* wqkvb = (bf16*)alloc((size_t)3 * DMODEL * DMODEL * 2);
  bf16* wg1b  = (bf16*)alloc((size_t)HD * DMODEL * 2);
  bf16* wg2b  = (bf16*)alloc((size_t)DMODEL * HD * 2);
  bf16* woutb = (bf16*)alloc((size_t)DMODEL * DMODEL * 2);
  bf16* qb    = (bf16*)alloc((size_t)M * DMODEL * 2);
  bf16* kb    = (bf16*)alloc((size_t)M * DMODEL * 2);
  bf16* vb    = (bf16*)alloc((size_t)M * DMODEL * 2);
  float* gatef = (float*)alloc((size_t)M * DMODEL * 4);
  float* attnf = (float*)alloc((size_t)M * DMODEL * 4);
  bf16* yb    = (bf16*)alloc((size_t)M * DMODEL * 2);

  auto cast = [&](const float* i, bf16* o, size_t n) {
    cast_kernel<<<(unsigned)((n / 4 + 255) / 256), 256, 0, stream>>>(i, o, (int)n);
  };
  cast(x, xb, (size_t)M * DMODEL);
  cast(wqkv, wqkvb, (size_t)3 * DMODEL * DMODEL);
  cast(wg1, wg1b, (size_t)HD * DMODEL);
  cast(wg2, wg2b, (size_t)DMODEL * HD);
  cast(wout, woutb, (size_t)DMODEL * DMODEL);

  dim3 gq(M / 128, (3 * DMODEL) / 128);
  gemm_qkv_kernel<<<gq, 256, 0, stream>>>(xb, wqkvb, qb, kb, vb, DMODEL);

  gate_kernel<<<M / 128, 256, 0, stream>>>(xb, wg1b, bg1, wg2b, bg2, gatef);

  dim3 ga(BQ * NH, NSEQ / 16 / 8);
  attn_kernel<<<ga, 256, 0, stream>>>(qb, kb, vb, attnf);

  ln_kernel<<<M, 256, 0, stream>>>(attnf, gatef, gamma, beta, yb);

  dim3 go(M / 128, DMODEL / 128);
  gemm_out_kernel<<<go, 256, 0, stream>>>(yb, woutb, out, DMODEL, DMODEL);
}
